// MultiHeadedAttention_3470333575652
// MI455X (gfx1250) — matrix operand order repaired and checked by reading, not compiled, not run on hardware
//
#include <hip/hip_runtime.h>
#include <hip/hip_bf16.h>
#include <math.h>

// CDNA5 / gfx1250, wave32. WMMA f32 = bf16 x bf16 + f32, 16x16x32.
typedef __attribute__((ext_vector_type(16))) __bf16 bf16x16;
typedef __attribute__((ext_vector_type(8)))  float  f32x8;

#define NEGV (-1e10f)   // log2-domain mask value (only relative magnitude matters)

static __device__ inline f32x8 wmma_bf16(bf16x16 a, bf16x16 b, f32x8 c) {
  return __builtin_amdgcn_wmma_f32_16x16x32_bf16(false, a, false, b, (short)0, c,
                                                 false, false);
}

// ---- operand loaders -------------------------------------------------------
// A operand (16x32, MxK), row-major source, leading dim ld (elements).
// Layout: lane (hi,m): elems 0..15 = K(hi*16 .. +15), row m.
static __device__ inline bf16x16 load_a_bf16(const __bf16* src, int ld, int rbase,
                                             int kbase, int lane) {
  int m = lane & 15, hi = lane >> 4;
  const __bf16* p = src + (size_t)(rbase + m) * ld + kbase + hi * 16;
  bf16x16 a;
#pragma unroll
  for (int i = 0; i < 8; ++i) { a[i] = p[i]; a[i + 8] = p[8 + i]; }
  return a;
}
static __device__ inline bf16x16 load_a_f32(const float* src, int ld, int rbase,
                                            int kbase, int lane) {
  int m = lane & 15, hi = lane >> 4;
  const float* p = src + (size_t)(rbase + m) * ld + kbase + hi * 16;
  bf16x16 a;
#pragma unroll
  for (int i = 0; i < 8; ++i) { a[i] = (__bf16)p[i]; a[i + 8] = (__bf16)p[8 + i]; }
  return a;
}
// B operand (32x16, KxN) where B[k,n] = src[nbase+n][kbase+k]  (i.e. computes X @ src^T).
// Layout: lane (hi,n): elems 0..15 = K(hi*16 .. +15), column n.
static __device__ inline bf16x16 load_b_f32(const float* src, int ld, int nbase,
                                            int kbase, int lane) {
  int n = lane & 15, hi = lane >> 4;
  const float* p = src + (size_t)(nbase + n) * ld + kbase + hi * 16;
  bf16x16 b;
#pragma unroll
  for (int i = 0; i < 16; ++i) b[i] = (__bf16)p[i];
  return b;
}

// ---- kernel 1: QKV projection  Y = X @ W^T + b  (M=16384, N=K=256) ---------
// vtrans==0: out[bth][s][d] (Q,K layout). vtrans==1: out[bth][d][s] (V transposed).
__global__ void qkv_proj_kernel(const float* __restrict__ X, const float* __restrict__ W,
                                const float* __restrict__ bias, __bf16* __restrict__ out,
                                int vtrans) {
  int lane = threadIdx.x & 31;
  int wave = threadIdx.x >> 5;
  int tile = blockIdx.x * 8 + wave;   // 16384 tiles (1024 row-tiles x 16 col-tiles)
  int sbase = (tile >> 4) * 16;       // row (flattened b*t*s)
  int nbase = (tile & 15) * 16;       // output feature col
  f32x8 acc = {};
#pragma unroll
  for (int kb = 0; kb < 256; kb += 32) {
    bf16x16 a = load_a_f32(X, 256, sbase, kb, lane);
    bf16x16 b = load_b_f32(W, 256, nbase, kb, lane);
    acc = wmma_bf16(a, b, acc);
  }
  int n = lane & 15, hi = lane >> 4;
  float bval = bias[nbase + n];
  int col = nbase + n;                 // global feature = h*32 + d
  int h = col >> 5, d = col & 31;
  int bt = sbase >> 10;                // 16-row tiles never cross the S=1024 boundary
  int s0 = (sbase & 1023) + hi * 8;    // C layout: VGPR r -> row r + 8*hi
  if (vtrans == 0) {
    __bf16* o = out + ((size_t)(bt * 8 + h) * 1024) * 32 + d;
#pragma unroll
    for (int r = 0; r < 8; ++r)
      o[(size_t)(s0 + r) * 32] = (__bf16)(acc[r] + bval);
  } else {
    __bf16* o = out + ((size_t)((bt * 8 + h) * 32 + d)) * 1024 + s0;
#pragma unroll
    for (int r = 0; r < 8; ++r)
      o[r] = (__bf16)(acc[r] + bval);
  }
}

// ---- kernel 2: flash attention per (b,t,h), 16-query tile per wave ---------
// scores^T tiles (keys = rows -> within-lane softmax axis), online base-2
// softmax, P^T consumed directly as the B operand of  x^T = V^T @ P^T.
__global__ void flash_attn_kernel(const __bf16* __restrict__ Qbf,
                                  const __bf16* __restrict__ Kbf,
                                  const __bf16* __restrict__ Vt,
                                  const int* __restrict__ mask,
                                  __bf16* __restrict__ Xo) {
  int lane = threadIdx.x & 31;
  int wave = threadIdx.x >> 5;
  int w = blockIdx.x * 8 + wave;   // 8192 waves total
  int bth = w >> 6;                // 0..127  (b*t*H + h)
  int qt  = w & 63;                // 64 query tiles of 16
  int bt = bth >> 3;
  int h  = bth & 7;
  int L = mask[bt];                // valid length, uniform across the wave
  int qbase = qt * 16;
  int n = lane & 15, hi = lane >> 4;

  const __bf16* Qh = Qbf + (size_t)bth * 1024 * 32;
  const __bf16* Kh = Kbf + (size_t)bth * 1024 * 32;
  const __bf16* Vh = Vt  + (size_t)bth * 32 * 1024;

  // Q tile as B operand (queries are the N dim): held in registers for all iters
  bf16x16 qb;
  {
    const __bf16* p = Qh + (size_t)(qbase + n) * 32 + hi * 16;
#pragma unroll
    for (int i = 0; i < 16; ++i) qb[i] = p[i];
  }
  bool qvalid = (qbase + n) < L;

  // Fully-valid q-tile: keys >= L contribute exactly 0 -> truncate the scan.
  // Any invalid query row softmaxes to uniform over ALL 1024 keys (reference
  // semantics), so mixed/invalid tiles must scan everything. Wave-uniform.
  int kend = (qbase + 16 <= L) ? ((L + 31) & ~31) : 1024;

  const float c2 = 0.25505402497917844f;  // (1/sqrt(32)) * log2(e)
  float mrun = -INFINITY, lrun = 0.f;
  f32x8 acc0 = {}, acc1 = {};             // x^T: dims 0..15 / 16..31, lane=query

  for (int kb = 0; kb < kend; kb += 32) {
    bf16x16 ak0 = load_a_bf16(Kh, 32, kb,      0, lane);
    bf16x16 ak1 = load_a_bf16(Kh, 32, kb + 16, 0, lane);
    if (kb + 32 < kend) {  // prefetch next iteration's K/V tiles (global_prefetch_b8)
      __builtin_prefetch(Kh + (size_t)(kb + 32 + (lane & 15)) * 32, 0, 0);
      __builtin_prefetch(Vh + (size_t)(lane & 15) * 1024 + kb + 32, 0, 0);
      __builtin_prefetch(Vh + (size_t)(16 + (lane & 15)) * 1024 + kb + 32, 0, 0);
    }
    f32x8 z = {};
    f32x8 s0 = wmma_bf16(ak0, qb, z);   // scores^T keys kb..kb+15
    f32x8 s1 = wmma_bf16(ak1, qb, z);   // scores^T keys kb+16..kb+31
    // lane holds keys r+8*hi (s0) and 16+r+8*hi (s1) of query column n
    float v0[8], v1[8];
    float mloc = -INFINITY;
#pragma unroll
    for (int r = 0; r < 8; ++r) {
      int k0 = kb + r + 8 * hi;
      v0[r] = (qvalid && k0 < L)      ? s0[r] * c2 : NEGV;   // log2 domain
      v1[r] = (qvalid && k0 + 16 < L) ? s1[r] * c2 : NEGV;
      mloc = fmaxf(mloc, fmaxf(v0[r], v1[r]));
    }
    mloc = fmaxf(mloc, __shfl_xor(mloc, 16, 32));  // combine the two key half-sets
    float mnew  = fmaxf(mrun, mloc);
    float alpha = exp2f(mrun - mnew);
    float p0[8], p1[8], lsum = 0.f;
#pragma unroll
    for (int r = 0; r < 8; ++r) {
      p0[r] = exp2f(v0[r] - mnew);
      p1[r] = exp2f(v1[r] - mnew);
      lsum += p0[r] + p1[r];
    }
    lsum += __shfl_xor(lsum, 16, 32);
    lrun = lrun * alpha + lsum;
    mrun = mnew;
#pragma unroll
    for (int r = 0; r < 8; ++r) { acc0[r] *= alpha; acc1[r] *= alpha; }
    // Reshape P^T (C layout) into B operand: lane-lo needs keys 0..15 (own 0..7 +
    // partner 8..15), lane-hi needs keys 16..31 (partner 16..23 + own 24..31).
    float o0[8], o1[8];
#pragma unroll
    for (int r = 0; r < 8; ++r) {
      o0[r] = __shfl_xor(p0[r], 16, 32);
      o1[r] = __shfl_xor(p1[r], 16, 32);
    }
    bf16x16 pb;
#pragma unroll
    for (int i = 0; i < 8; ++i) {
      pb[i]     = (__bf16)((hi == 0) ? p0[i] : o1[i]);
      pb[i + 8] = (__bf16)((hi == 0) ? o0[i] : p1[i]);
    }
    // x^T += V^T @ P^T   (A = V^T rows d, K = 32 keys)
    bf16x16 av0 = load_a_bf16(Vh, 1024, 0,  kb, lane);
    bf16x16 av1 = load_a_bf16(Vh, 1024, 16, kb, lane);
    acc0 = wmma_bf16(av0, pb, acc0);
    acc1 = wmma_bf16(av1, pb, acc1);
  }
  float inv = 1.f / lrun;
  // write x[row = bt*1024 + q][h*32 + d];  acc0: d = r+8*hi, acc1: d = 16+r+8*hi
  __bf16* orow = Xo + ((size_t)(bt * 1024 + qbase + n)) * 256 + h * 32;
#pragma unroll
  for (int r = 0; r < 8; ++r) {
    orow[8 * hi + r]      = (__bf16)(acc0[r] * inv);
    orow[16 + 8 * hi + r] = (__bf16)(acc1[r] * inv);
  }
}

// ---- kernel 3: output projection  out = Xa @ Wo^T + bo  (f32 out) ----------
__global__ void out_proj_kernel(const __bf16* __restrict__ Xa, const float* __restrict__ Wo,
                                const float* __restrict__ bo, float* __restrict__ out) {
  int lane = threadIdx.x & 31;
  int wave = threadIdx.x >> 5;
  int tile = blockIdx.x * 8 + wave;
  int sbase = (tile >> 4) * 16;
  int nbase = (tile & 15) * 16;
  f32x8 acc = {};
#pragma unroll
  for (int kb = 0; kb < 256; kb += 32) {
    bf16x16 a = load_a_bf16(Xa, 256, sbase, kb, lane);
    bf16x16 b = load_b_f32(Wo, 256, nbase, kb, lane);
    acc = wmma_bf16(a, b, acc);
  }
  int n = lane & 15, hi = lane >> 4;
  float bval = bo[nbase + n];
  float* o = out + (size_t)(sbase + 8 * hi) * 256 + nbase + n;
#pragma unroll
  for (int r = 0; r < 8; ++r) o[(size_t)r * 256] = acc[r] + bval;
}

extern "C" void kernel_launch(void* const* d_in, const int* in_sizes, int n_in,
                              void* d_out, int out_size, void* d_ws, size_t ws_size,
                              hipStream_t stream) {
  const float* query = (const float*)d_in[0];
  const float* key   = (const float*)d_in[1];
  const float* value = (const float*)d_in[2];
  const int*   mask  = (const int*)d_in[3];
  const float* Wq = (const float*)d_in[4];
  const float* bq = (const float*)d_in[5];
  const float* Wk = (const float*)d_in[6];
  const float* bk = (const float*)d_in[7];
  const float* Wv = (const float*)d_in[8];
  const float* bv = (const float*)d_in[9];
  const float* Wo = (const float*)d_in[10];
  const float* bo = (const float*)d_in[11];
  float* out = (float*)d_out;

  char* ws = (char*)d_ws;
  __bf16* Qbf = (__bf16*)(ws);                      // [128][1024][32]  8 MB
  __bf16* Kbf = (__bf16*)(ws + ((size_t)8  << 20)); // [128][1024][32]  8 MB
  __bf16* Vt  = (__bf16*)(ws + ((size_t)16 << 20)); // [128][32][1024]  8 MB
  __bf16* Xa  = (__bf16*)(ws + ((size_t)24 << 20)); // [16384][256]     8 MB

  dim3 blk(256);  // 8 waves per WGP-resident block
  qkv_proj_kernel<<<2048, blk, 0, stream>>>(query, Wq, bq, Qbf, 0);
  qkv_proj_kernel<<<2048, blk, 0, stream>>>(key,   Wk, bk, Kbf, 0);
  qkv_proj_kernel<<<2048, blk, 0, stream>>>(value, Wv, bv, Vt, 1);
  flash_attn_kernel<<<1024, blk, 0, stream>>>(Qbf, Kbf, Vt, mask, Xa);
  out_proj_kernel<<<2048, blk, 0, stream>>>(Xa, Wo, bo, out);
}
